// PosLSTM_76115410420290
// MI455X (gfx1250) — matrix operand order repaired and checked by reading, not compiled, not run on hardware
//
#include <hip/hip_runtime.h>

// ---------------------------------------------------------------------------
// PosLSTM for MI455X (gfx1250, wave32, WMMA).
//   B=64, P=16, T=256, I=256, H=512, O=5, V=50000
// Stage 1: pack weights/emb to f16 (WMMA fragment layout for B operands).
// Stage 2: gather-fused GEMM  pre[t*64+b, g*512+h] = emb[x] @ Wx  (+bias)
//          M=16384, K=4096, N=2048, f16 WMMA, f32 accum.
// Stage 3: persistent single-WGP recurrence, h in LDS(f16), c in VGPRs,
//          4 gates per step via WMMA against packed Wh.
// ---------------------------------------------------------------------------

typedef __attribute__((ext_vector_type(16))) _Float16 v16h;
typedef __attribute__((ext_vector_type(8)))  _Float16 v8h;
typedef __attribute__((ext_vector_type(8)))  float    v8f;

#define BB   64
#define PP   16
#define TT   256
#define II   256
#define HH   512
#define OO   5
#define VV   50000
#define K1   4096          // P*I
#define N1   2048          // 4*H
#define KT1  128           // K1/32
#define KT2  16            // H/32

// ---------------- Stage 1 kernels ----------------

__global__ void emb_to_f16(const float* __restrict__ emb, _Float16* __restrict__ out, int n) {
  int i = blockIdx.x * blockDim.x + threadIdx.x;
  if (i >= n) return;
  float v = (i < II) ? 0.0f : emb[i];   // padding_idx=0 -> zero row 0
  out[i] = (_Float16)v;
}

// Pack a [K,512]-per-gate weight set (4 gates -> N=2048) into WMMA B-fragment
// order. Fragment f = nt*KT + kt holds 512 halves: lane l owns 16 contiguous
// halves: N = nt*16 + (l&15), K = kt*32 + (l>>4)*8 + (j&7) + (j>>3)*16.
__global__ void pack_w(const float* __restrict__ w0, const float* __restrict__ w1,
                       const float* __restrict__ w2, const float* __restrict__ w3,
                       _Float16* __restrict__ out, int ktBits, int ktMask, int total) {
  int idx = blockIdx.x * blockDim.x + threadIdx.x;
  if (idx >= total) return;
  int f   = idx >> 9;
  int rem = idx & 511;
  int l   = rem >> 4;
  int j   = rem & 15;
  int nt  = f >> ktBits;
  int kt  = f & ktMask;
  int n   = nt * 16 + (l & 15);
  int k   = kt * 32 + (l >> 4) * 8 + (j & 7) + (j >> 3) * 16;
  int g   = n >> 9;
  int h   = n & 511;
  const float* w = (g == 0) ? w0 : (g == 1) ? w1 : (g == 2) ? w2 : w3;
  out[idx] = (_Float16)w[(size_t)k * 512 + h];   // [.,512] row-major: works for Wx([P*I,H]) and Wh([H,H])
}

__global__ void pack_bias(const float* __restrict__ b0, const float* __restrict__ b1,
                          const float* __restrict__ b2, const float* __restrict__ b3,
                          float* __restrict__ out) {
  int i = blockIdx.x * blockDim.x + threadIdx.x;
  if (i >= N1) return;
  int g = i >> 9;
  const float* b = (g == 0) ? b0 : (g == 1) ? b1 : (g == 2) ? b2 : b3;
  out[i] = b[i & 511];
}

// ---------------- Stage 2: gather-fused big GEMM ----------------
// grid = (N1/128, TT); block = 128 (4 waves). Each block handles one t
// (64 rows = all b) x 128 columns; each wave: 64x32 output via 4x2 WMMA tiles.
__global__ __launch_bounds__(128) void gemm_x(const int* __restrict__ x,
                                              const _Float16* __restrict__ embh,
                                              const _Float16* __restrict__ wpack,
                                              const float* __restrict__ bias,
                                              float* __restrict__ pre) {
  const int lane = threadIdx.x & 31;
  const int wave = threadIdx.x >> 5;
  const int r    = lane & 15;       // A-row / C-col within tile
  const int gl   = lane >> 4;       // half-wave group (K phase / C-row phase)
  const int t    = blockIdx.y;                       // one t per block (64 rows)
  const int nBase = blockIdx.x * 128 + wave * 32;
  const int nt0   = nBase >> 4;

  v8f acc[4][2];
#pragma unroll
  for (int mi = 0; mi < 4; ++mi)
#pragma unroll
    for (int ni = 0; ni < 2; ++ni)
#pragma unroll
      for (int e = 0; e < 8; ++e) acc[mi][ni][e] = 0.0f;

  for (int p = 0; p < PP; ++p) {
    int rows[4];
#pragma unroll
    for (int mi = 0; mi < 4; ++mi) {
      int b = mi * 16 + r;                      // row within the 64-row block
      rows[mi] = x[((size_t)b * PP + p) * TT + t];
    }
#pragma unroll
    for (int kk = 0; kk < 8; ++kk) {            // 8 k-tiles of 32 inside one p
      const int kt = p * 8 + kk;
      v16h a[4];
#pragma unroll
      for (int mi = 0; mi < 4; ++mi) {
        // ISA A-layout (16-bit, 16x32): lane group gl holds K = gl*8+{0..7} and +16
        const _Float16* ap = embh + (size_t)rows[mi] * II + kk * 32 + gl * 8;
        v8h lo = *(const v8h*)(ap);
        v8h hi = *(const v8h*)(ap + 16);
#pragma unroll
        for (int j = 0; j < 8; ++j) { a[mi][j] = lo[j]; a[mi][j + 8] = hi[j]; }
      }
#pragma unroll
      for (int ni = 0; ni < 2; ++ni) {
        const _Float16* bp = wpack + ((size_t)(nt0 + ni) * KT1 + kt) * 512 + lane * 16;
        __builtin_prefetch(bp + 512, 0, 3);     // next-kt fragment -> global_prefetch
        v16h bf = *(const v16h*)bp;
#pragma unroll
        for (int mi = 0; mi < 4; ++mi)
          acc[mi][ni] = __builtin_amdgcn_wmma_f32_16x16x32_f16(
              false, a[mi], false, bf, (short)0, acc[mi][ni], false, false);
      }
    }
  }

  // C layout: element e -> row = e + gl*8, col = r (within 16x16 tile)
#pragma unroll
  for (int mi = 0; mi < 4; ++mi)
#pragma unroll
    for (int ni = 0; ni < 2; ++ni)
#pragma unroll
      for (int e = 0; e < 8; ++e) {
        int row = t * 64 + mi * 16 + gl * 8 + e;       // m = t*64 + b
        int col = nBase + ni * 16 + r;
        pre[(size_t)row * N1 + col] = acc[mi][ni][e] + bias[col];
      }
}

// ---------------- Stage 3: persistent recurrence ----------------
// One workgroup, 1024 threads = 32 waves. Wave w owns h-columns [w*16, w*16+16).
// Per step: gates[64, w*16..] for all 4 gates via WMMA (A = h from LDS,
// B = packed Wh from L2), cell update in registers, h written back to LDS.
__global__ __launch_bounds__(1024) void lstm_rec(const float* __restrict__ pre,
                                                 const _Float16* __restrict__ whpack,
                                                 const float* __restrict__ wlin,
                                                 const float* __restrict__ blin,
                                                 float* __restrict__ out) {
  __shared__ _Float16 hsh[BB * HH];               // 64 KB, h state (f16)
  const int tid  = threadIdx.x;
  const int lane = tid & 31;
  const int wave = tid >> 5;
  const int r    = lane & 15;
  const int gl   = lane >> 4;
  const int hc   = wave * 16 + r;                  // owned h column

  for (int i = tid; i < BB * HH; i += 1024) hsh[i] = (_Float16)0.0f;

  v8f creg[4];
#pragma unroll
  for (int mi = 0; mi < 4; ++mi)
#pragma unroll
    for (int e = 0; e < 8; ++e) creg[mi][e] = 0.0f;

  __syncthreads();

  for (int t = 0; t < TT; ++t) {
    v8f acc[4][4];                                 // [gate][mtile]
#pragma unroll
    for (int g = 0; g < 4; ++g)
#pragma unroll
      for (int mi = 0; mi < 4; ++mi)
#pragma unroll
        for (int e = 0; e < 8; ++e) acc[g][mi][e] = 0.0f;

    for (int kt = 0; kt < KT2; ++kt) {
      v16h a[4];
#pragma unroll
      for (int mi = 0; mi < 4; ++mi) {
        const _Float16* ap = &hsh[(size_t)(mi * 16 + r) * HH + kt * 32 + gl * 8];
        v8h lo = *(const v8h*)(ap);
        v8h hi = *(const v8h*)(ap + 16);
#pragma unroll
        for (int j = 0; j < 8; ++j) { a[mi][j] = lo[j]; a[mi][j + 8] = hi[j]; }
      }
#pragma unroll
      for (int g = 0; g < 4; ++g) {
        const _Float16* bp = whpack + ((size_t)(g * 32 + wave) * KT2 + kt) * 512 + lane * 16;
        v16h bf = *(const v16h*)bp;
#pragma unroll
        for (int mi = 0; mi < 4; ++mi)
          acc[g][mi] = __builtin_amdgcn_wmma_f32_16x16x32_f16(
              false, a[mi], false, bf, (short)0, acc[g][mi], false, false);
      }
    }

    float hn[4][8];
#pragma unroll
    for (int mi = 0; mi < 4; ++mi) {
#pragma unroll
      for (int e = 0; e < 8; ++e) {
        int row = mi * 16 + gl * 8 + e;            // = b
        size_t base = ((size_t)t * 64 + row) * N1;
        float pf = acc[0][mi][e] + pre[base +        hc];
        float pi = acc[1][mi][e] + pre[base +  512 + hc];
        float pg = acc[2][mi][e] + pre[base + 1024 + hc];
        float po = acc[3][mi][e] + pre[base + 1536 + hc];
        float fg = 1.0f / (1.0f + __expf(-pf));
        float ig = 1.0f / (1.0f + __expf(-pi));
        float gg = tanhf(pg);
        float og = 1.0f / (1.0f + __expf(-po));
        float c  = fg * creg[mi][e] + ig * gg;
        creg[mi][e] = c;
        float h  = og * tanhf(c);
        hn[mi][e] = h;
        if (t == TT - 1) {
          out[320 +         (size_t)row * HH + hc] = h;   // h_t (f32)
          out[320 + 32768 + (size_t)row * HH + hc] = c;   // c_t (f32)
        }
      }
    }

    __syncthreads();                               // all LDS reads of h done
#pragma unroll
    for (int mi = 0; mi < 4; ++mi)
#pragma unroll
      for (int e = 0; e < 8; ++e) {
        int row = mi * 16 + gl * 8 + e;
        hsh[(size_t)row * HH + hc] = (_Float16)hn[mi][e];
      }
    __syncthreads();                               // h visible for next step
  }

  // out = h_T @ W_lin + b_lin  (read f32 h_T written above; visible after barrier)
  if (tid < BB * OO) {
    int b = tid / OO, o = tid % OO;
    float s = blin[o];
    const float* hrow = out + 320 + (size_t)b * HH;
    for (int k = 0; k < HH; ++k) s += hrow[k] * wlin[k * OO + o];
    out[b * OO + o] = s;
  }
}

// ---------------- host launcher ----------------

extern "C" void kernel_launch(void* const* d_in, const int* in_sizes, int n_in,
                              void* d_out, int out_size, void* d_ws, size_t ws_size,
                              hipStream_t stream) {
  (void)in_sizes; (void)n_in; (void)out_size; (void)ws_size;
  const int*   x    = (const int*)  d_in[0];
  const float* emb  = (const float*)d_in[1];
  const float* wfx  = (const float*)d_in[2];
  const float* wfh  = (const float*)d_in[3];
  const float* bf   = (const float*)d_in[4];
  const float* wix  = (const float*)d_in[5];
  const float* wih  = (const float*)d_in[6];
  const float* bi   = (const float*)d_in[7];
  const float* wgx  = (const float*)d_in[8];
  const float* wgh  = (const float*)d_in[9];
  const float* bg   = (const float*)d_in[10];
  const float* wox  = (const float*)d_in[11];
  const float* woh  = (const float*)d_in[12];
  const float* bo   = (const float*)d_in[13];
  const float* wlin = (const float*)d_in[14];
  const float* blin = (const float*)d_in[15];

  // workspace layout (256B aligned slabs)
  char* ws = (char*)d_ws;
  const size_t embElems = (size_t)VV * II;                  // 12,800,000 halves
  const size_t wxElems  = (size_t)N1 * K1;                  //  8,388,608 halves
  const size_t whElems  = (size_t)N1 * HH;                  //  1,048,576 halves
  size_t off = 0;
  auto take = [&](size_t bytes) { char* p = ws + off; off += (bytes + 255) & ~(size_t)255; return p; };
  _Float16* embh   = (_Float16*)take(embElems * 2);
  _Float16* wxpack = (_Float16*)take(wxElems * 2);
  _Float16* whpack = (_Float16*)take(whElems * 2);
  float*    biasA  = (float*)   take((size_t)N1 * 4);
  float*    pre    = (float*)   take((size_t)TT * BB * N1 * 4);   // 134 MB

  // Stage 1
  {
    int n = (int)embElems;
    emb_to_f16<<<(n + 255) / 256, 256, 0, stream>>>(emb, embh, n);
  }
  pack_w<<<(int)((wxElems + 255) / 256), 256, 0, stream>>>(wfx, wix, wgx, wox,
                                                           wxpack, 7, 127, (int)wxElems);
  pack_w<<<(int)((whElems + 255) / 256), 256, 0, stream>>>(wfh, wih, wgh, woh,
                                                           whpack, 4, 15, (int)whElems);
  pack_bias<<<(N1 + 255) / 256, 256, 0, stream>>>(bf, bi, bg, bo, biasA);

  // Stage 2: M=16384 (t-major, 64 b-rows per block), N=2048
  gemm_x<<<dim3(N1 / 128, TT), 128, 0, stream>>>(x, embh, wxpack, biasA, pre);

  // Stage 3: single persistent workgroup
  lstm_rec<<<1, 1024, 0, stream>>>(pre, whpack, wlin, blin, (float*)d_out);
}
